// MultiHeadAttention_1142461301048
// MI455X (gfx1250) — compile-verified, matrix operand order checked by reading, not run on hardware
//
#include <hip/hip_runtime.h>
#include <hip/hip_bf16.h>
#include <stdint.h>

// MI455X (gfx1250) multi-head attention:
//   mask bit-pack (ballot) + 3x projection GEMM (f32 -> f16 WMMA, f32 acc) ->
//   flash attention (WMMA, double-buffered async global->LDS K/V tiles,
//   WMMA row-sum softmax) -> output projection GEMM.  Needs 66 MB of d_ws.

typedef __attribute__((ext_vector_type(16))) _Float16 v16h;
typedef __attribute__((ext_vector_type(8)))  _Float16 v8h;
typedef __attribute__((ext_vector_type(8)))  float    v8f;
typedef __attribute__((ext_vector_type(4)))  float    v4f;

static constexpr int EMB = 1024, HEADS = 16, BATCH = 4, SEQ = 2048, HD = 64;

union Frag16 { v16h v; v8h h[2]; };

__device__ __forceinline__ v16h pack2(v8h lo, v8h hi) {
  Frag16 f; f.h[0] = lo; f.h[1] = hi; return f.v;
}

// ---- CDNA5 async global->LDS copy (ASYNCcnt path) via inline asm ----------
#if defined(__gfx1250__)
#define USE_ASYNC_LDS 1
#else
#define USE_ASYNC_LDS 0
#endif

#if USE_ASYNC_LDS
__device__ __forceinline__ void async_copy16(const void* g, void* l) {
  // LDS byte offset = value of the AS(3) pointer (low 32 bits of generic ptr)
  uint32_t loff = (uint32_t)(size_t)(__attribute__((address_space(3))) char*)l;
  asm volatile("global_load_async_to_lds_b128 %0, %1, off"
               :: "v"(loff), "v"(g) : "memory");
}
template<int N>
__device__ __forceinline__ void wait_async() {
  asm volatile("s_wait_asynccnt %0" :: "i"(N) : "memory");
}
#endif

// ---------------------------------------------------------------------------
// Mask bit-pack: mask[B,1,S,S] int32 -> bits[B*S][S/32] uint32 via wave ballot
// ---------------------------------------------------------------------------
__global__ __launch_bounds__(256)
void maskpack_kernel(const int* __restrict__ mask, uint32_t* __restrict__ mb)
{
  const int lane = threadIdx.x & 31;
  const int gw   = blockIdx.x * 8 + (threadIdx.x >> 5);   // global wave id
  const int row  = gw >> 6, word = gw & 63;               // 64 words per row
  int m = mask[(size_t)row * SEQ + word * 32 + lane];
  unsigned long long bal = __ballot(m != 0);
  if (lane == 0) mb[(size_t)row * 64 + word] = (uint32_t)bal;
}

// ---------------------------------------------------------------------------
// GEMM: C[M,N] = A[M,K] * W[N,K]^T + bias   (M=8192, N=K=1024)
// MODE 0: f16 store [B,H,S,D] (Q,K)   MODE 2: f16 store [B,H,D,S] (V^T)
// MODE 4: f32 store [M,N] (final output)
// ---------------------------------------------------------------------------
template<int MODE, bool A_HALF>
__global__ __launch_bounds__(256)
void gemm_kernel(const void* __restrict__ Av, const float* __restrict__ W,
                 const float* __restrict__ bias, void* __restrict__ dstv)
{
  __shared__ _Float16 lds_a[128 * 32];
  __shared__ _Float16 lds_w[64 * 32];

  const int tid  = threadIdx.x;
  const int lane = tid & 31, wave = tid >> 5;
  const int l16  = lane & 15, half = lane >> 4;
  const int wm   = wave >> 1, wn = wave & 1;

  const int n_base = blockIdx.x * 64;
  const int m_base = blockIdx.y * 128;

  v8f acc[2][2] = {};

  const int arow = tid >> 1, aseg = tid & 1;
  const int wrow = tid >> 2, wseg = tid & 3;

  v4f a_f[4]; v8h a_h[2]; v4f w_f[2];          // one K-step lookahead

  auto load_stage = [&](int k0) {
    if (A_HALF) {
      const _Float16* Ah = (const _Float16*)Av;
      const v8h* src = (const v8h*)(Ah + (size_t)(m_base + arow) * EMB + k0 + aseg * 16);
      a_h[0] = src[0]; a_h[1] = src[1];
    } else {
      const float* Af = (const float*)Av;
      const v4f* src = (const v4f*)(Af + (size_t)(m_base + arow) * EMB + k0 + aseg * 16);
      a_f[0] = src[0]; a_f[1] = src[1]; a_f[2] = src[2]; a_f[3] = src[3];
    }
    const v4f* wsrc = (const v4f*)(W + (size_t)(n_base + wrow) * EMB + k0 + wseg * 8);
    w_f[0] = wsrc[0]; w_f[1] = wsrc[1];
  };

  auto store_stage = [&]() {
    if (A_HALF) {
      *(v8h*)&lds_a[arow * 32 + aseg * 16 + 0] = a_h[0];
      *(v8h*)&lds_a[arow * 32 + aseg * 16 + 8] = a_h[1];
    } else {
      v8h h0, h1;
      #pragma unroll
      for (int i = 0; i < 4; i++) {
        h0[i] = (_Float16)a_f[0][i]; h0[i + 4] = (_Float16)a_f[1][i];
        h1[i] = (_Float16)a_f[2][i]; h1[i + 4] = (_Float16)a_f[3][i];
      }
      *(v8h*)&lds_a[arow * 32 + aseg * 16 + 0] = h0;
      *(v8h*)&lds_a[arow * 32 + aseg * 16 + 8] = h1;
    }
    v8h hw;
    #pragma unroll
    for (int i = 0; i < 4; i++) { hw[i] = (_Float16)w_f[0][i]; hw[i + 4] = (_Float16)w_f[1][i]; }
    *(v8h*)&lds_w[wrow * 32 + wseg * 8] = hw;
  };

  load_stage(0);
  for (int k0 = 0; k0 < EMB; k0 += 32) {
    store_stage();
    __syncthreads();
    if (k0 + 32 < EMB) load_stage(k0 + 32);   // prefetch next tile during WMMA

    v16h afrag[2], bfrag[2];
    #pragma unroll
    for (int ms = 0; ms < 2; ms++) {
      int r = wm * 32 + ms * 16 + l16;
      afrag[ms] = pack2(*(const v8h*)&lds_a[r * 32 + half * 8],
                        *(const v8h*)&lds_a[r * 32 + 16 + half * 8]);
    }
    #pragma unroll
    for (int ns = 0; ns < 2; ns++) {
      int n = wn * 32 + ns * 16 + l16;
      bfrag[ns] = pack2(*(const v8h*)&lds_w[n * 32 + half * 8],
                        *(const v8h*)&lds_w[n * 32 + 16 + half * 8]);
    }
    #pragma unroll
    for (int ms = 0; ms < 2; ms++)
      #pragma unroll
      for (int ns = 0; ns < 2; ns++)
        acc[ms][ns] = __builtin_amdgcn_wmma_f32_16x16x32_f16(
            false, afrag[ms], false, bfrag[ns], (short)0, acc[ms][ns], false, false);
    __syncthreads();
  }

  #pragma unroll
  for (int ms = 0; ms < 2; ms++) {
    #pragma unroll
    for (int ns = 0; ns < 2; ns++) {
      int ng = n_base + wn * 32 + ns * 16 + l16;
      float bvv = bias[ng];
      #pragma unroll
      for (int r = 0; r < 8; r++) {
        int mg = m_base + wm * 32 + ms * 16 + half * 8 + r;
        float val = acc[ms][ns][r] + bvv;
        if (MODE == 4) {
          ((float*)dstv)[(size_t)mg * EMB + ng] = val;
        } else {
          int bb = mg >> 11, s = mg & (SEQ - 1);
          int hh = ng >> 6,  d = ng & (HD - 1);
          _Float16* dst = (_Float16*)dstv;
          if (MODE == 2)
            dst[(((size_t)bb * HEADS + hh) * HD + d) * SEQ + s] = (_Float16)val;   // V^T
          else
            dst[(((size_t)bb * HEADS + hh) * SEQ + s) * HD + d] = (_Float16)val;   // Q,K
        }
      }
    }
  }
}

// ---------------------------------------------------------------------------
// Flash attention: workgroup = 8 waves x 16 query rows of one (b,h).
// Key blocks of 32 double-buffered in LDS via async copies. WMMA scores,
// WMMA row-sum for the softmax normalizer, shfl_xor row-max only.
// ---------------------------------------------------------------------------
__global__ __launch_bounds__(256)
void attention_kernel(const _Float16* __restrict__ Qh, const _Float16* __restrict__ Kh,
                      const _Float16* __restrict__ Vt, const uint32_t* __restrict__ mbits,
                      _Float16* __restrict__ Oh)
{
  __shared__ _Float16 lds_k[2][32 * 64];   // [buf][key][d]   2x4 KB
  __shared__ _Float16 lds_v[2][64 * 32];   // [buf][d][key]   2x4 KB
  __shared__ _Float16 lds_p[8][16 * 32];   // per-wave P        8 KB

  const int tid  = threadIdx.x;
  const int lane = tid & 31, wave = tid >> 5;
  const int l16  = lane & 15, half = lane >> 4;

  const int bh = blockIdx.y;                 // b*H + h
  const int b = bh >> 4, h = bh & 15;
  const int q0 = (blockIdx.x * 8 + wave) * 16;

  const _Float16* Qp = Qh + (size_t)bh * SEQ * HD;
  const _Float16* Kp = Kh + (size_t)bh * SEQ * HD;
  const _Float16* Vp = Vt + (size_t)bh * HD * SEQ;
  const uint32_t* mbp = mbits + (size_t)b * SEQ * 64;   // per-row 64 words

  const int krow = tid >> 3, kseg = tid & 7;
  const int vrow = tid >> 2, vseg = tid & 3;

#if USE_ASYNC_LDS
  auto stage = [&](int kb, int buf) {   // 2 async b128 ops per thread
    async_copy16(Kp + (size_t)(kb + krow) * HD + kseg * 8,
                 &lds_k[buf][krow * 64 + kseg * 8]);
    async_copy16(Vp + (size_t)vrow * SEQ + kb + vseg * 8,
                 &lds_v[buf][vrow * 32 + vseg * 8]);
  };
#else
  auto stage = [&](int kb, int buf) {
    *(v8h*)&lds_k[buf][krow * 64 + kseg * 8] =
        *(const v8h*)(Kp + (size_t)(kb + krow) * HD + kseg * 8);
    *(v8h*)&lds_v[buf][vrow * 32 + vseg * 8] =
        *(const v8h*)(Vp + (size_t)vrow * SEQ + kb + vseg * 8);
  };
#endif

  // Q fragments (reused for all keys)
  v16h aq[2];
  #pragma unroll
  for (int dh = 0; dh < 2; dh++) {
    const _Float16* p = Qp + (size_t)(q0 + l16) * HD + dh * 32 + half * 8;
    aq[dh] = pack2(*(const v8h*)p, *(const v8h*)(p + 16));
  }

  // all-ones B fragment: wmma(P, ones) -> row sums of P in C layout
  v16h ones;
  #pragma unroll
  for (int i = 0; i < 16; i++) ones[i] = (_Float16)1.0f;

  v8f o[4] = {};
  float mrow[8], lrow[8];
  #pragma unroll
  for (int r = 0; r < 8; r++) { mrow[r] = -3.0e38f; lrow[r] = 0.f; }
  const float scale = 0.125f;   // 1/sqrt(64)

  _Float16* pt = &lds_p[wave][0];

  stage(0, 0);
  int ibuf = 0;
  for (int kb = 0; kb < SEQ; kb += 32, ibuf ^= 1) {
#if USE_ASYNC_LDS
    if (kb + 32 < SEQ) { stage(kb + 32, ibuf ^ 1); wait_async<2>(); }
    else                { wait_async<0>(); }
#else
    if (kb + 32 < SEQ) stage(kb + 32, ibuf ^ 1);
#endif
    __syncthreads();

    const _Float16* kt = &lds_k[ibuf][0];
    const _Float16* vt = &lds_v[ibuf][0];

    // ---- scores: load all 4 K fragments first so ds_loads pipeline ----
    v16h bk[2][2];                       // [nh][dh]
    #pragma unroll
    for (int nh = 0; nh < 2; nh++)
      #pragma unroll
      for (int dh = 0; dh < 2; dh++) {
        const _Float16* p = &kt[(nh * 16 + l16) * 64 + dh * 32 + half * 8];
        bk[nh][dh] = pack2(*(const v8h*)p, *(const v8h*)(p + 16));
      }
    v8f s0 = {}, s1 = {};
    s0 = __builtin_amdgcn_wmma_f32_16x16x32_f16(false, aq[0], false, bk[0][0], (short)0, s0, false, false);
    s0 = __builtin_amdgcn_wmma_f32_16x16x32_f16(false, aq[1], false, bk[0][1], (short)0, s0, false, false);
    s1 = __builtin_amdgcn_wmma_f32_16x16x32_f16(false, aq[0], false, bk[1][0], (short)0, s1, false, false);
    s1 = __builtin_amdgcn_wmma_f32_16x16x32_f16(false, aq[1], false, bk[1][1], (short)0, s1, false, false);

    // ---- scale + packed-bit mask + row max (shfl tree) ----
    float corr8[8];
    #pragma unroll
    for (int r = 0; r < 8; r++) {
      int row = q0 + half * 8 + r;
      uint32_t w = mbp[(size_t)row * 64 + (kb >> 5)];
      s0[r] = ((w >> l16) & 1u)        ? -1e9f : s0[r] * scale;
      s1[r] = ((w >> (16 + l16)) & 1u) ? -1e9f : s1[r] * scale;
    }
    #pragma unroll
    for (int r = 0; r < 8; r++) {
      float v = fmaxf(s0[r], s1[r]);
      v = fmaxf(v, __shfl_xor(v, 1, 32));
      v = fmaxf(v, __shfl_xor(v, 2, 32));
      v = fmaxf(v, __shfl_xor(v, 4, 32));
      v = fmaxf(v, __shfl_xor(v, 8, 32));
      float mnew = fmaxf(mrow[r], v);
      corr8[r] = __expf(mrow[r] - mnew);
      mrow[r] = mnew;
      // P tile straight to LDS as f16 (A-layout reload below)
      pt[(half * 8 + r) * 32 + l16]      = (_Float16)__expf(s0[r] - mnew);
      pt[(half * 8 + r) * 32 + 16 + l16] = (_Float16)__expf(s1[r] - mnew);
      #pragma unroll
      for (int t = 0; t < 4; t++) o[t][r] *= corr8[r];
    }
    asm volatile("s_wait_dscnt 0" ::: "memory");   // same-wave LDS RAW fence

    v16h pa = pack2(*(const v8h*)&pt[l16 * 32 + half * 8],
                    *(const v8h*)&pt[l16 * 32 + 16 + half * 8]);

    // ---- row sums via WMMA: (P @ ones)[m][n] = sum_k P[m][k] ----
    v8f rs = {};
    rs = __builtin_amdgcn_wmma_f32_16x16x32_f16(false, pa, false, ones, (short)0, rs, false, false);
    #pragma unroll
    for (int r = 0; r < 8; r++) lrow[r] = lrow[r] * corr8[r] + rs[r];

    // ---- O(16x64) += P(16x32) @ V(32x64); load all 4 V frags first ----
    v16h bv[4];
    #pragma unroll
    for (int t = 0; t < 4; t++) {
      const _Float16* p = &vt[(t * 16 + l16) * 32 + half * 8];
      bv[t] = pack2(*(const v8h*)p, *(const v8h*)(p + 16));
    }
    #pragma unroll
    for (int t = 0; t < 4; t++)
      o[t] = __builtin_amdgcn_wmma_f32_16x16x32_f16(false, pa, false, bv[t], (short)0, o[t], false, false);
    __syncthreads();   // tile consumed; its buffer may be overwritten
  }

  // ---- normalize, store to Oh[B,S,E] f16 ----
  #pragma unroll
  for (int r = 0; r < 8; r++) {
    float inv = 1.f / lrow[r];
    int row = q0 + half * 8 + r;
    #pragma unroll
    for (int t = 0; t < 4; t++) {
      int d = t * 16 + l16;
      Oh[((size_t)b * SEQ + row) * EMB + h * HD + d] = (_Float16)(o[t][r] * inv);
    }
  }
}

// ---------------------------------------------------------------------------
extern "C" void kernel_launch(void* const* d_in, const int* in_sizes, int n_in,
                              void* d_out, int out_size, void* d_ws, size_t ws_size,
                              hipStream_t stream)
{
  (void)in_sizes; (void)n_in; (void)out_size; (void)ws_size;
  const float* q    = (const float*)d_in[0];
  const float* k    = (const float*)d_in[1];
  const float* v    = (const float*)d_in[2];
  const int*   mask = (const int*)  d_in[3];
  const float* Wq   = (const float*)d_in[4];
  const float* bq   = (const float*)d_in[5];
  const float* Wk   = (const float*)d_in[6];
  const float* bk   = (const float*)d_in[7];
  const float* Wv   = (const float*)d_in[8];
  const float* bv   = (const float*)d_in[9];
  const float* Wo   = (const float*)d_in[10];
  const float* bo   = (const float*)d_in[11];

  // workspace: 4 x 16 MB f16 tensors + 2 MB mask bits = 66 MB
  const size_t elems = (size_t)BATCH * HEADS * SEQ * HD;   // 8M halves
  _Float16* Qh = (_Float16*)d_ws;
  _Float16* Kh = Qh + elems;
  _Float16* Vt = Kh + elems;     // [B,H,D,S]
  _Float16* Oh = Vt + elems;     // [B,S,E]
  uint32_t* Mb = (uint32_t*)(Oh + elems);   // [B*S][64] bit-packed mask

  hipLaunchKernelGGL(maskpack_kernel, dim3((BATCH * SEQ * 64) / 8), dim3(256), 0, stream, mask, Mb);

  dim3 gblk(256), ggrid(EMB / 64, (BATCH * SEQ) / 128);
  hipLaunchKernelGGL((gemm_kernel<0, false>), ggrid, gblk, 0, stream, (const void*)q, Wq, bq, (void*)Qh);
  hipLaunchKernelGGL((gemm_kernel<0, false>), ggrid, gblk, 0, stream, (const void*)k, Wk, bk, (void*)Kh);
  hipLaunchKernelGGL((gemm_kernel<2, false>), ggrid, gblk, 0, stream, (const void*)v, Wv, bv, (void*)Vt);

  dim3 agrid(SEQ / 16 / 8, BATCH * HEADS);
  hipLaunchKernelGGL(attention_kernel, agrid, dim3(256), 0, stream, Qh, Kh, Vt, Mb, Oh);

  hipLaunchKernelGGL((gemm_kernel<4, true>), ggrid, gblk, 0, stream, (const void*)Oh, Wo, bo, d_out);
}
